// TemporalTransformerEncoderLayer_21328807592600
// MI455X (gfx1250) — matrix operand order repaired and checked by reading, not compiled, not run on hardware
//
#include <hip/hip_runtime.h>

// ---------------------------------------------------------------------------
// MI455X / gfx1250 implementation of TemporalTransformerEncoderLayer
// (multi-scale deformable attention). wave32, WMMA bf16 matrix cores,
// Tensor Data Mover for bf16 A-panel staging in the output projection GEMM.
// ---------------------------------------------------------------------------

typedef float  v8f     __attribute__((ext_vector_type(8)));
typedef __bf16 bf16x4  __attribute__((ext_vector_type(4)));
typedef __bf16 bf16x8  __attribute__((ext_vector_type(8)));
typedef __bf16 bf16x16 __attribute__((ext_vector_type(16)));
typedef unsigned int u32x4 __attribute__((ext_vector_type(4)));
typedef int i32x4 __attribute__((ext_vector_type(4)));
typedef int i32x8 __attribute__((ext_vector_type(8)));

#define DMODEL 256
#define LDA 40   // padded LDS strides (bytes: 80, multiple of 16 -> b128 aligned)
#define LDB 40

// ROCm 7.2 clang-22: 5-arg tensor_load_to_lds builtin.
// amdgpu-toolchain (therock): 6-arg builtin + TDM header (unique to it).
#if defined(__has_include)
#if __has_include(<hip/amd_detail/amd_gfx1250_TDM.h>)
#define TDM_6ARG 1
#else
#define TDM_6ARG 0
#endif
#else
#define TDM_6ARG 0
#endif

// ---------------------------------------------------------------------------
// TDM: DMA a contiguous run of `nelem` bf16 elements from global to LDS.
// Descriptor bitfields per CDNA5 ISA sec 8.3/8.4:
//   g0: count=1 | lds_addr | global_addr[56:0] | type=2
//   g1: data_size=1 (2B), tensor_dim0 = tile_dim0 = nelem, stride = nelem (1-D)
// Issued by one wave; completion via s_wait_tensorcnt (TENSORcnt).
// ---------------------------------------------------------------------------
__device__ __forceinline__ void tdm_load_1d(const void* gsrc, unsigned int lds_addr,
                                            unsigned int nelem)
{
    unsigned long long ga = (unsigned long long)(uintptr_t)gsrc;
    u32x4 g0;
    g0[0] = 1u;                                                  // count=1 (valid user D#)
    g0[1] = lds_addr;                                            // LDS byte address
    g0[2] = (unsigned int)ga;                                    // global_addr[31:0]
    g0[3] = (unsigned int)((ga >> 32) & 0x01FFFFFFu) | (2u << 30); // [56:32] + type=2
    i32x8 g1;
    g1[0] = (int)(1u << 16);                                     // data_size=1 -> 2 bytes
    g1[1] = (int)((nelem & 0xFFFFu) << 16);                      // tensor_dim0[15:0]
    g1[2] = (int)((nelem >> 16) & 0xFFFFu);                      // tensor_dim0[31:16]
    g1[3] = (int)((nelem & 0xFFFFu) << 16);                      // tile_dim0 (<=65535)
    g1[4] = 0;                                                   // tile_dim1/2 unused
    g1[5] = (int)nelem;                                          // tensor_dim0_stride lo32
    g1[6] = 0;
    g1[7] = 0;
    i32x4 gz = {0, 0, 0, 0};
#if TDM_6ARG
    i32x8 gz8 = {0, 0, 0, 0, 0, 0, 0, 0};
    __builtin_amdgcn_tensor_load_to_lds(g0, g1, gz, gz, gz8, 0);
#else
    __builtin_amdgcn_tensor_load_to_lds(g0, g1, gz, gz, 0);
#endif
}

// ---------------------------------------------------------------------------
// Weight packing: f32 -> bf16, concat [W_off | W_attn] into 480-col matrix
// ---------------------------------------------------------------------------
__global__ __launch_bounds__(256)
void pack_weights_kernel(const float* __restrict__ Woff, const float* __restrict__ Wattn,
                         const float* __restrict__ Wval, const float* __restrict__ Wout,
                         const float* __restrict__ boff, const float* __restrict__ battn,
                         __bf16* __restrict__ woa, __bf16* __restrict__ wval,
                         __bf16* __restrict__ wout, float* __restrict__ boa)
{
    int i = blockIdx.x * 256 + threadIdx.x;
    const int nWoa = 4 * 256 * 480;
    const int nWv  = 4 * 256 * 256;
    if (i < nWoa) {
        int lvl = i / (256 * 480);
        int rem = i - lvl * (256 * 480);
        int k = rem / 480;
        int c = rem - k * 480;
        float v = (c < 320) ? Woff[((size_t)lvl * 256 + k) * 320 + c]
                            : Wattn[((size_t)lvl * 256 + k) * 160 + (c - 320)];
        woa[i] = (__bf16)v;
        return;
    }
    i -= nWoa;
    if (i < nWv) { wval[i] = (__bf16)Wval[i]; return; }
    i -= nWv;
    if (i < nWv) { wout[i] = (__bf16)Wout[i]; return; }
    i -= nWv;
    if (i < 4 * 480) {
        int lvl = i / 480, c = i - lvl * 480;
        boa[i] = (c < 320) ? boff[lvl * 320 + c] : battn[lvl * 160 + (c - 320)];
    }
}

// ---------------------------------------------------------------------------
// bf16 WMMA GEMM:  Y[M,N] = A[M,256] * W[256,N] + bias[N]
//  AMODE 0: A = (src + pos) f32, converted to bf16 while staging into LDS
//  AMODE 1: A = bf16 array; entire 64x256 A-panel staged ONCE by the
//           Tensor Data Mover (tensor_load_to_lds + s_wait_tensorcnt)
//  OUTBF 0: f32 output  / 1: bf16 output
// Block: 256 threads = 8 waves. Tile 64(M) x 32(N), K-step 32, 8 steps.
// Wave wv: rows 16*(wv&3), cols 16*(wv>>2) -> one 16x16 f32 accumulator.
// ---------------------------------------------------------------------------
template <int AMODE, int OUTBF>
__global__ __launch_bounds__(256)
void gemm_kernel(const float* __restrict__ srcA, const float* __restrict__ posA,
                 const __bf16* __restrict__ Abf,
                 const __bf16* __restrict__ W, const float* __restrict__ bias,
                 float* __restrict__ outF, __bf16* __restrict__ outB, int N)
{
    constexpr int ASTRIDE = (AMODE == 1) ? DMODEL : LDA;
    __shared__ __bf16 ldsA[64 * ASTRIDE];  // AMODE0: [row][k-step]; AMODE1: full panel
    __shared__ __bf16 ldsB[32 * LDB];      // transposed: [col][k]

    const int t  = threadIdx.x;
    const int m0 = blockIdx.y * 64;
    const int n0 = blockIdx.x * 32;

    // A-tile loader mapping (AMODE 0): 64 rows x 32 k, 8 bf16 per thread
    const int ar   = t >> 2;
    const int aseg = (t & 3) * 8;
    // B-tile loader mapping: 32 k x 32 cols, 4 bf16 per thread (scatter-transpose)
    const int bk = t >> 3;
    const int bc = (t & 7) * 4;

    const int lane = t & 31;
    const int wv   = t >> 5;
    const int wm   = wv & 3;
    const int wn   = wv >> 2;
    const int lr   = lane & 15;
    const int kh   = lane >> 4;

    if (AMODE == 1) {
        // One wave issues the TDM DMA of the whole contiguous 64x256 bf16 panel.
        if (wv == 0) {
            tdm_load_1d(Abf + (size_t)m0 * DMODEL,
                        (unsigned int)(uintptr_t)&ldsA[0], 64 * DMODEL);
        }
    }

    v8f acc = {0.f, 0.f, 0.f, 0.f, 0.f, 0.f, 0.f, 0.f};

    // B 32x16 frag: lane group kh holds K 8kh..8kh+7 & 16+8kh..; transposed LDS makes K contiguous
    const __bf16* bFrag0 = &ldsB[(wn * 16 + lr) * LDB + kh * 8];
    const __bf16* bFrag1 = bFrag0 + 16;

    #pragma unroll
    for (int kt = 0; kt < 8; ++kt) {
        const int k0 = kt * 32;
        if (AMODE == 0) {
            const float* ps = srcA + (size_t)(m0 + ar) * DMODEL + k0 + aseg;
            const float* pp = posA + (size_t)(m0 + ar) * DMODEL + k0 + aseg;
            float4 s0 = *(const float4*)ps;
            float4 s1 = *(const float4*)(ps + 4);
            float4 p0 = *(const float4*)pp;
            float4 p1 = *(const float4*)(pp + 4);
            bf16x8 av;
            av[0] = (__bf16)(s0.x + p0.x); av[1] = (__bf16)(s0.y + p0.y);
            av[2] = (__bf16)(s0.z + p0.z); av[3] = (__bf16)(s0.w + p0.w);
            av[4] = (__bf16)(s1.x + p1.x); av[5] = (__bf16)(s1.y + p1.y);
            av[6] = (__bf16)(s1.z + p1.z); av[7] = (__bf16)(s1.w + p1.w);
            *(bf16x8*)&ldsA[ar * LDA + aseg] = av;
            if (kt < 7) { __builtin_prefetch(ps + 32, 0, 3); __builtin_prefetch(pp + 32, 0, 3); }
        }
        {
            const __bf16* pb = W + (size_t)(k0 + bk) * N + n0 + bc;
            bf16x4 b4 = *(const bf16x4*)pb;
            ldsB[(bc + 0) * LDB + bk] = b4[0];
            ldsB[(bc + 1) * LDB + bk] = b4[1];
            ldsB[(bc + 2) * LDB + bk] = b4[2];
            ldsB[(bc + 3) * LDB + bk] = b4[3];
            if (kt < 7) __builtin_prefetch(pb + (size_t)32 * N, 0, 3);
        }
        if (AMODE == 1 && kt == 0) {
            // Waves with no outstanding tensor op see TENSORcnt==0 (no-op wait);
            // the issuing wave blocks until the DMA has landed in LDS. The
            // workgroup barrier below then publishes the panel to all waves.
            __builtin_amdgcn_s_wait_tensorcnt(0);
        }
        __syncthreads();

        // ISA 16-bit A 16x32 layout: lane<16 holds K 0..7 & 16..23; lane>=16: 8..15 & 24..31
        const __bf16* aBase =
            &ldsA[(wm * 16 + lr) * ASTRIDE + (AMODE == 1 ? k0 : 0) + kh * 8];
        union { bf16x16 v; bf16x8 h[2]; } ua, ub;
        ua.h[0] = *(const bf16x8*)aBase;
        ua.h[1] = *(const bf16x8*)(aBase + 16);
        ub.h[0] = *(const bf16x8*)bFrag0;
        ub.h[1] = *(const bf16x8*)bFrag1;
        acc = __builtin_amdgcn_wmma_f32_16x16x32_bf16(
                  false, ua.v, false, ub.v, (short)0, acc, false, false);
        __syncthreads();
    }

    // C/D 16x16 f32 layout: VGPR r -> M = r + 8*(lane>=16), N = lane&15
    const int rowb = m0 + wm * 16 + kh * 8;
    const int col  = n0 + wn * 16 + lr;
    const float bv = bias[col];
    #pragma unroll
    for (int r = 0; r < 8; ++r) {
        size_t o = (size_t)(rowb + r) * (size_t)N + col;
        float v = acc[r] + bv;
        if (OUTBF) outB[o] = (__bf16)v;
        else       outF[o] = v;
    }
}

// ---------------------------------------------------------------------------
// Deformable sampling: one wave32 per (query, head); lane = channel d (0..31).
// oa[L,480]: cols 0..319 offsets ((h*5+l)*4+p)*2+xy, 320..479 attn logits h*20+l*4+p.
// val bf16 [L,256] head-major; slice l occupies rows [l*s*s, (l+1)*s*s).
// ---------------------------------------------------------------------------
__global__ __launch_bounds__(256)
void sampling_kernel(const float* __restrict__ oa, const __bf16* __restrict__ valbf,
                     const float* __restrict__ ref, __bf16* __restrict__ sampled, int s)
{
    const int q = blockIdx.x;
    const int h = threadIdx.x >> 5;
    const int d = threadIdx.x & 31;
    const float* oaq = oa + (size_t)q * 480;

    // softmax over 20 logits (wave-uniform values, scalarizes well)
    float lg[20];
    float mx = -1e30f;
    #pragma unroll
    for (int i = 0; i < 20; ++i) { lg[i] = oaq[320 + h * 20 + i]; mx = fmaxf(mx, lg[i]); }
    float ssum = 0.f;
    #pragma unroll
    for (int i = 0; i < 20; ++i) { lg[i] = __expf(lg[i] - mx); ssum += lg[i]; }
    const float inv = 1.f / ssum;

    const int HW = s * s;
    const float fs = (float)s;
    float acc = 0.f;

    for (int l = 0; l < 5; ++l) {
        const __bf16* vl = valbf + (size_t)l * HW * 256 + h * 32 + d;
        const float rx = ref[((size_t)q * 5 + l) * 2 + 0];
        const float ry = ref[((size_t)q * 5 + l) * 2 + 1];
        #pragma unroll
        for (int p = 0; p < 4; ++p) {
            const int oc = ((h * 5 + l) * 4 + p) * 2;
            // (ref + off/s)*s - 0.5 == ref*s + off - 0.5  (H == W == s)
            const float x = rx * fs + oaq[oc]     - 0.5f;
            const float y = ry * fs + oaq[oc + 1] - 0.5f;
            const float x0f = floorf(x), y0f = floorf(y);
            const float fx = x - x0f, fy = y - y0f;
            const int x0 = (int)x0f, y0 = (int)y0f;
            const float aw = lg[l * 4 + p] * inv;
            #pragma unroll
            for (int dy = 0; dy < 2; ++dy) {
                #pragma unroll
                for (int dx = 0; dx < 2; ++dx) {
                    const int xi = x0 + dx, yi = y0 + dy;
                    if (xi >= 0 && xi < s && yi >= 0 && yi < s) {  // wave-uniform branch
                        const float w = (dx ? fx : 1.f - fx) * (dy ? fy : 1.f - fy);
                        const float g = (float)vl[(size_t)(yi * s + xi) * 256];
                        acc += aw * w * g;
                    }
                }
            }
        }
    }
    sampled[(size_t)q * 256 + h * 32 + d] = (__bf16)acc;
}

// ---------------------------------------------------------------------------
// Residual + LayerNorm: wave32 per row, 8 channels per lane, shfl_xor reduce.
// out = LN((src+pos) + y2) * g + b
// ---------------------------------------------------------------------------
__global__ __launch_bounds__(256)
void ln_kernel(const float* __restrict__ src, const float* __restrict__ pos,
               const float* __restrict__ y2, const float* __restrict__ g,
               const float* __restrict__ b, float* __restrict__ out, int L)
{
    const int lane = threadIdx.x & 31;
    const int wv   = threadIdx.x >> 5;
    const int q    = blockIdx.x * 8 + wv;
    if (q >= L) return;
    const float* ps = src + (size_t)q * 256;
    const float* pp = pos + (size_t)q * 256;
    const float* py = y2  + (size_t)q * 256;

    float v[8];
    float sum = 0.f;
    #pragma unroll
    for (int j = 0; j < 8; ++j) {
        int c = lane * 8 + j;
        v[j] = ps[c] + pp[c] + py[c];
        sum += v[j];
    }
    #pragma unroll
    for (int off = 16; off; off >>= 1) sum += __shfl_xor(sum, off, 32);
    const float mu = sum * (1.f / 256.f);
    float var = 0.f;
    #pragma unroll
    for (int j = 0; j < 8; ++j) { float dv = v[j] - mu; var += dv * dv; }
    #pragma unroll
    for (int off = 16; off; off >>= 1) var += __shfl_xor(var, off, 32);
    const float rstd = rsqrtf(var * (1.f / 256.f) + 1e-5f);

    float* po = out + (size_t)q * 256;
    #pragma unroll
    for (int j = 0; j < 8; ++j) {
        int c = lane * 8 + j;
        po[c] = (v[j] - mu) * rstd * g[c] + b[c];
    }
}

// ---------------------------------------------------------------------------
// Orchestration. Input dict order:
// 0:src0 1:pos0 2:ref0 3:mask0 4:src1 ... 15:mask3
// 16:W_off 17:b_off 18:W_attn 19:b_attn 20:W_val 21:b_val 22:W_out 23:b_out
// 24:ln_g 25:ln_b 26:spatial_shapes (static; masks all-false -> ignored)
// ---------------------------------------------------------------------------
extern "C" void kernel_launch(void* const* d_in, const int* in_sizes, int n_in,
                              void* d_out, int out_size, void* d_ws, size_t ws_size,
                              hipStream_t stream)
{
    (void)in_sizes; (void)n_in; (void)out_size; (void)ws_size;

    const float* src[4] = {(const float*)d_in[0], (const float*)d_in[4],
                           (const float*)d_in[8], (const float*)d_in[12]};
    const float* pos[4] = {(const float*)d_in[1], (const float*)d_in[5],
                           (const float*)d_in[9], (const float*)d_in[13]};
    const float* ref[4] = {(const float*)d_in[2], (const float*)d_in[6],
                           (const float*)d_in[10], (const float*)d_in[14]};
    const float* W_off  = (const float*)d_in[16];
    const float* b_off  = (const float*)d_in[17];
    const float* W_attn = (const float*)d_in[18];
    const float* b_attn = (const float*)d_in[19];
    const float* W_val  = (const float*)d_in[20];
    const float* b_val  = (const float*)d_in[21];
    const float* W_out  = (const float*)d_in[22];
    const float* b_out  = (const float*)d_in[23];
    const float* ln_g   = (const float*)d_in[24];
    const float* ln_b   = (const float*)d_in[25];
    float* out = (float*)d_out;

    // Workspace layout (256B-aligned offsets), level scratch sized for level 0:
    char* ws = (char*)d_ws;
    __bf16* woa  = (__bf16*)(ws + 0);          // 4*256*480*2  =   983040
    __bf16* wval = (__bf16*)(ws + 983040);     // 4*256*256*2  =   524288
    __bf16* wout = (__bf16*)(ws + 1507328);    // 4*256*256*2  =   524288
    float*  boa  = (float*) (ws + 2031616);    // 4*480*4      =     7680
    float*  oa   = (float*) (ws + 2039296);    // 20480*480*4  = 39321600
    __bf16* valb = (__bf16*)(ws + 41360896);   // 20480*256*2  = 10485760
    __bf16* samp = (__bf16*)(ws + 51846656);   // 20480*256*2  = 10485760
    float*  y2   = (float*) (ws + 62332416);   // 20480*256*4  = 20971520
                                               // total        = 83303936 B

    {
        const int total = 4*256*480 + 4*256*256 + 4*256*256 + 4*480;
        pack_weights_kernel<<<(total + 255) / 256, 256, 0, stream>>>(
            W_off, W_attn, W_val, W_out, b_off, b_attn, woa, wval, wout, boa);
    }

    const int sArr[4] = {64, 32, 16, 8};
    int qoff = 0;
    for (int lvl = 0; lvl < 4; ++lvl) {
        const int s = sArr[lvl];
        const int L = 5 * s * s;                 // 20480 / 5120 / 1280 / 320

        // offsets + attn logits: [L,480] f32
        gemm_kernel<0, 0><<<dim3(480 / 32, L / 64), 256, 0, stream>>>(
            src[lvl], pos[lvl], nullptr,
            woa + (size_t)lvl * 256 * 480, boa + lvl * 480, oa, nullptr, 480);

        // values: [L,256] bf16 (halves the gather traffic; fits in L2)
        gemm_kernel<0, 1><<<dim3(256 / 32, L / 64), 256, 0, stream>>>(
            src[lvl], pos[lvl], nullptr,
            wval + (size_t)lvl * 256 * 256, b_val + lvl * 256, nullptr, valb, 256);

        // deformable attention sampling: wave per (q, head)
        sampling_kernel<<<dim3(L), 256, 0, stream>>>(oa, valb, ref[lvl], samp, s);

        // output projection: [L,256] f32, A-panel staged by the Tensor Data Mover
        gemm_kernel<1, 0><<<dim3(256 / 32, L / 64), 256, 0, stream>>>(
            nullptr, nullptr, samp,
            wout + (size_t)lvl * 256 * 256, b_out + lvl * 256, y2, nullptr, 256);

        // residual + layernorm into output slice
        ln_kernel<<<dim3(L / 8), 256, 0, stream>>>(
            src[lvl], pos[lvl], y2, ln_g, ln_b, out + (size_t)qoff * 256, L);

        qoff += L;
    }
}
